// TLSTM_43980465111645
// MI455X (gfx1250) — compile-verified, matrix operand order checked by reading, not run on hardware
//
#include <hip/hip_runtime.h>
#include <stdint.h>

#define BDIM 128
#define SDIM 1024
#define IDIM 128
#define HDIM 256
#define BT   16          // batch tile per workgroup
#define HSTR 264         // padded LDS row stride (elements) for h/c (528B, 16B aligned)
#define XSTR 136         // padded LDS row stride (elements) for x   (272B)

typedef __attribute__((ext_vector_type(16))) __bf16 v16bf;
typedef __attribute__((ext_vector_type(8)))  __bf16 v8bf;
typedef __attribute__((ext_vector_type(8)))  float  v8f;

// ---------------- device helpers ----------------

__device__ __forceinline__ v8f wmma_bf16(v16bf a, v16bf b, v8f c) {
  // D = A(16x32 bf16) * B(32x16 bf16) + C(16x16 f32)
  return __builtin_amdgcn_wmma_f32_16x16x32_bf16(
      /*neg_a=*/false, a, /*neg_b=*/false, b,
      /*c_mod=*/(short)0, c, /*reuse_a=*/false, /*reuse_b=*/false);
}

// Per-lane fragment load for the ISA 16-bit A/B layout: two 16B chunks at +0/+32B.
__device__ __forceinline__ v16bf frag(const __bf16* p) {
  v8bf lo = *(const v8bf*)(p);        // K pairs 0..7   (this lane's half)
  v8bf hi = *(const v8bf*)(p + 16);   // K pairs 16..23 (this lane's half)
  return __builtin_shufflevector(lo, hi, 0,1,2,3,4,5,6,7,8,9,10,11,12,13,14,15);
}

__device__ __forceinline__ void async_b128(uint32_t lds_off, uint64_t gaddr) {
  asm volatile("global_load_async_to_lds_b128 %0, %1, off"
               :: "v"(lds_off), "v"(gaddr) : "memory");
}
__device__ __forceinline__ void wait_async0() {
  asm volatile("s_wait_asynccnt 0" ::: "memory");
}

__device__ __forceinline__ float fsig(float x) {      // 1/(1+e^-x)
  return __builtin_amdgcn_rcpf(1.0f + __builtin_amdgcn_exp2f(-1.442695041f * x));
}
__device__ __forceinline__ float ftanh(float x) {
  return 2.0f * fsig(2.0f * x) - 1.0f;
}

// ---------------- phase 0: weights f32 -> bf16 into ws ----------------
// ws layout (elements): Ui,Uf,Uo,Uc,Wd (65536 each) then Wi,Wf,Wo,Wc (32768 each)

__global__ void convert_weights(const float* __restrict__ Ui, const float* __restrict__ Uf,
                                const float* __restrict__ Uo, const float* __restrict__ Uc,
                                const float* __restrict__ Wd, const float* __restrict__ Wi,
                                const float* __restrict__ Wf, const float* __restrict__ Wo,
                                const float* __restrict__ Wc, __bf16* __restrict__ dst)
{
  int g = blockIdx.x * blockDim.x + threadIdx.x;
  const int UT  = 5 * 65536;            // 327680
  const int TOT = UT + 4 * 32768;       // 458752
  if (g >= TOT) return;
  float v;
  if (g < UT) {
    int seg = g >> 16, idx = g & 65535;
    const float* s = (seg == 0) ? Ui : (seg == 1) ? Uf : (seg == 2) ? Uo
                     : (seg == 3) ? Uc : Wd;
    v = s[idx];
  } else {
    int g2 = g - UT;
    int seg = g2 >> 15, idx = g2 & 32767;
    const float* s = (seg == 0) ? Wi : (seg == 1) ? Wf : (seg == 2) ? Wo : Wc;
    v = s[idx];
  }
  dst[g] = (__bf16)v;
}

// ---------------- phase 1: persistent TLSTM scan ----------------
// 8 workgroups (one per 16-batch tile), 512 threads = 16 waves.
// Wave w owns output columns [16w,16w+16): 56 v_wmma per wave per step.
// Weights stream from L2 every step (0.9MB hot set); LICM defeated via
// an opaque per-iteration integer offset (keeps global_load + saddr form).

__global__ void __launch_bounds__(512)
tlstm_scan(const float* __restrict__ x, const float* __restrict__ ts,
           const float* __restrict__ bI, const float* __restrict__ bF,
           const float* __restrict__ bO, const float* __restrict__ bC,
           const float* __restrict__ bD,
           const __bf16* __restrict__ w, float* __restrict__ out)
{
  __shared__ __bf16 h_lds[BT * HSTR];        // bf16 h_t (A-operand for U gemms)
  __shared__ __bf16 c_lds[BT * HSTR];        // bf16 c_t (A-operand for Wd gemm)
  __shared__ __bf16 x_bf [BT * XSTR];        // bf16 x_t (A-operand for W gemms)
  __shared__ float  x_f32[2][BT * IDIM];     // async double buffer for raw x_t

  const int tid   = threadIdx.x;
  const int lane  = tid & 31;
  const int wv    = tid >> 5;                // 0..15
  const int half  = lane >> 4;               // 0|1
  const int mrow  = lane & 15;               // A-row / B-column within tile
  const int klane = half * 8;                // element offset for this half-wave
  const int nn    = wv * 16 + mrow;          // global output column
  const int b0    = blockIdx.x * BT;

  // per-lane element offsets into each bf16 weight segment (base applied in-loop)
  const size_t oUi =      0 + (size_t)nn * HDIM + klane;
  const size_t oUf =  65536 + (size_t)nn * HDIM + klane;
  const size_t oUo = 131072 + (size_t)nn * HDIM + klane;
  const size_t oUc = 196608 + (size_t)nn * HDIM + klane;
  const size_t oWd = 262144 + (size_t)nn * HDIM + klane;
  const size_t oWi = 327680 + (size_t)nn * IDIM + klane;
  const size_t oWf = 360448 + (size_t)nn * IDIM + klane;
  const size_t oWo = 393216 + (size_t)nn * IDIM + klane;
  const size_t oWc = 425984 + (size_t)nn * IDIM + klane;

  const __bf16* hA = h_lds + mrow * HSTR + klane;
  const __bf16* cA = c_lds + mrow * HSTR + klane;
  const __bf16* xA = x_bf  + mrow * XSTR + klane;

  const float biasI = bI[nn], biasF = bF[nn], biasO = bO[nn],
              biasC = bC[nn], biasD = bD[nn];

  // x staging: thread tid moves 4 consecutive f32 (16B) per step
  const int e  = tid * 4;
  const int mx = e >> 7;                     // batch row within tile
  const int kx = e & 127;                    // input feature
  const float* xsrc  = x + ((size_t)(b0 + mx) * SDIM) * IDIM + kx;
  float*  xdstA   = &x_f32[0][tid * 4];
  float*  xdstB   = &x_f32[1][tid * 4];
  __bf16* xbf_dst = &x_bf[mx * XSTR + kx];

  const float* tsrc = ts + (size_t)(b0 + mrow) * SDIM;

  // single output base; row m = r + 8*half handled via immediate r*S*H offsets
  float* outBase = out + ((size_t)(b0 + klane) * SDIM) * HDIM + nn;

  // zero initial h/c state in LDS
  for (int i2 = tid; i2 < BT * HSTR; i2 += 512) {
    h_lds[i2] = (__bf16)0.0f;
    c_lds[i2] = (__bf16)0.0f;
  }

  // prefetch x[t=0] (async -> LDS), convert to bf16, prefetch x[t=1]
  async_b128((uint32_t)(uintptr_t)xdstA, (uint64_t)(uintptr_t)xsrc);
  wait_async0();
  {
    float4 v = *(const float4*)xdstA;
    xbf_dst[0] = (__bf16)v.x; xbf_dst[1] = (__bf16)v.y;
    xbf_dst[2] = (__bf16)v.z; xbf_dst[3] = (__bf16)v.w;
  }
  async_b128((uint32_t)(uintptr_t)xdstB, (uint64_t)(uintptr_t)(xsrc + IDIM));
  __syncthreads();

  float c_reg[8], h_out[8];
  #pragma unroll
  for (int r = 0; r < 8; ++r) { c_reg[r] = 0.0f; h_out[r] = 0.0f; }

  for (int t = 0; t < SDIM; ++t) {
    // Opaque zero offset, renewed every iteration: the weight base keeps its
    // global-address-space provenance (-> global_load_b128, LOADcnt only,
    // saddr + loop-invariant voffset + folded immediates), but LICM cannot
    // hoist the 112 in-loop B-fragment loads (no scratch spills).
    size_t zo = 0;
    asm volatile("" : "+s"(zo));
    const __bf16* wb = w + zo;

    // T_t = 1/log(ts + e + 1e-6), per batch row (held by lanes 0..15/16..31)
    float tsv  = tsrc[t];
    float Tval = __builtin_amdgcn_rcpf(
        0.69314718056f * __builtin_amdgcn_logf(tsv + 2.7183f + 1e-6f));

    v8f aI = {}, aF = {}, aO = {}, aC = {}, aD = {};

    // input-side GEMMs fused into gate accumulators (K = 128)
    #pragma unroll
    for (int kc = 0; kc < IDIM / 32; ++kc) {
      v16bf ax = frag(xA + kc * 32);
      aI = wmma_bf16(ax, frag(wb + oWi + kc * 32), aI);
      aF = wmma_bf16(ax, frag(wb + oWf + kc * 32), aF);
      aO = wmma_bf16(ax, frag(wb + oWo + kc * 32), aO);
      aC = wmma_bf16(ax, frag(wb + oWc + kc * 32), aC);
    }
    // recurrent GEMMs (K = 256)
    #pragma unroll
    for (int kc = 0; kc < HDIM / 32; ++kc) {
      v16bf ah = frag(hA + kc * 32);
      aI = wmma_bf16(ah, frag(wb + oUi + kc * 32), aI);
      aF = wmma_bf16(ah, frag(wb + oUf + kc * 32), aF);
      aO = wmma_bf16(ah, frag(wb + oUo + kc * 32), aO);
      aC = wmma_bf16(ah, frag(wb + oUc + kc * 32), aC);
      v16bf ac = frag(cA + kc * 32);
      aD = wmma_bf16(ac, frag(wb + oWd + kc * 32), aD);
    }

    // elementwise gate math on the C/D tile (row m = r + 8*half, col nn)
    #pragma unroll
    for (int r = 0; r < 8; ++r) {
      float Tm = __shfl(Tval, r + klane, 32);          // broadcast T[m]
      float dg = ftanh(aD[r] + biasD);                 // C_ST
      float cr = c_reg[r] + dg * (Tm - 1.0f);          // decayed cell
      float ig = fsig(aI[r] + biasI);
      float fg = fsig(aF[r] + biasF);
      float og = fsig(aO[r] + biasO);
      float cc = ftanh(aC[r] + biasC);
      float cn = fg * cr + ig * cc;
      float hn = og * ftanh(cn);
      c_reg[r] = cn;
      h_out[r] = hn;
      // streaming output: NT hint so 128MB of outputs doesn't evict weights in L2
      __builtin_nontemporal_store(
          hn, outBase + (size_t)r * SDIM * HDIM + (size_t)t * HDIM);
    }

    __syncthreads();   // everyone done reading h_lds/c_lds/x_bf for step t

    // publish bf16 h/c for next step's K-reduction
    #pragma unroll
    for (int r = 0; r < 8; ++r) {
      int m = r + klane;
      h_lds[m * HSTR + nn] = (__bf16)h_out[r];
      c_lds[m * HSTR + nn] = (__bf16)c_reg[r];
    }

    // stage x for step t+1 (async buffer landed during this step's GEMMs)
    if (t + 1 < SDIM) {
      wait_async0();
      const float* src = ((t + 1) & 1) ? xdstB : xdstA;
      float4 v = *(const float4*)src;
      xbf_dst[0] = (__bf16)v.x; xbf_dst[1] = (__bf16)v.y;
      xbf_dst[2] = (__bf16)v.z; xbf_dst[3] = (__bf16)v.w;
      if (t + 2 < SDIM) {
        float* dst = ((t + 2) & 1) ? xdstB : xdstA;
        async_b128((uint32_t)(uintptr_t)dst,
                   (uint64_t)(uintptr_t)(xsrc + (size_t)(t + 2) * IDIM));
      }
    }
    __syncthreads();   // publish h/c/x for next step
  }

  // final state outputs: h_n[1,B,H], c_n[1,B,H]
  float* hn_out = out + (size_t)BDIM * SDIM * HDIM;
  float* cn_out = hn_out + (size_t)BDIM * HDIM;
  #pragma unroll
  for (int r = 0; r < 8; ++r) {
    int m = r + klane;
    hn_out[(size_t)(b0 + m) * HDIM + nn] = h_out[r];
    cn_out[(size_t)(b0 + m) * HDIM + nn] = c_reg[r];
  }
}

// ---------------- host launch ----------------

extern "C" void kernel_launch(void* const* d_in, const int* in_sizes, int n_in,
                              void* d_out, int out_size, void* d_ws, size_t ws_size,
                              hipStream_t stream)
{
  const float* inputs = (const float*)d_in[0];
  const float* tstamp = (const float*)d_in[1];
  const float* Wi_w = (const float*)d_in[2];
  const float* Wi_b = (const float*)d_in[3];
  const float* Ui_w = (const float*)d_in[4];
  const float* Wf_w = (const float*)d_in[5];
  const float* Wf_b = (const float*)d_in[6];
  const float* Uf_w = (const float*)d_in[7];
  const float* Wo_w = (const float*)d_in[8];
  const float* Wo_b = (const float*)d_in[9];
  const float* Uo_w = (const float*)d_in[10];
  const float* Wc_w = (const float*)d_in[11];
  const float* Wc_b = (const float*)d_in[12];
  const float* Uc_w = (const float*)d_in[13];
  const float* Wd_w = (const float*)d_in[14];
  const float* Wd_b = (const float*)d_in[15];

  __bf16* wbf = (__bf16*)d_ws;  // 458752 bf16 = ~0.9MB, stays hot in 192MB L2

  convert_weights<<<(458752 + 255) / 256, 256, 0, stream>>>(
      Ui_w, Uf_w, Uo_w, Uc_w, Wd_w, Wi_w, Wf_w, Wo_w, Wc_w, wbf);

  tlstm_scan<<<BDIM / BT, 512, 0, stream>>>(
      inputs, tstamp, Wi_b, Wf_b, Wo_b, Wc_b, Wd_b, wbf, (float*)d_out);
}
